// NeuralEvaluatorModel_20375324852345
// MI455X (gfx1250) — compile-verified, matrix operand order checked by reading, not run on hardware
//
#include <hip/hip_runtime.h>
#include <hip/hip_bf16.h>

// ============================================================================
// Stacked-LSTM (8 layers x 4096 steps, shared carry) for MI455X / gfx1250.
//
// 32768 strictly sequential LSTM cells, each a 4096x1124 bf16 GEMV done with
// v_wmma_f32_16x16x32_bf16 (B = vector in column 0; columns 1-15 carry
// replicated finite garbage that never reaches column 0 of D -> no zeroing,
// no EXEC juggling in the inner loop).
//
// Partitioning: 64 WGs; WG g owns h/c indices [16g, 16g+16). Its 8 waves
// cover the 4 gate blocks x 2 K-halves (dependent WMMA chain = 9 per
// accumulator). c stays WG-local; one grid-wide sync per cell broadcasts h.
// Weights live in L2 (75 MB bf16 packed); WGP-scope prefetch pulls upcoming
// A-tiles into the near cache. Workspace requirement: ~80 MB.
// ============================================================================

#define HIDDEN   1024
#define LAYERS   8
#define LETTERS  100
#define T_TOTAL  4096
#define KPAD     128      // W_ih K padded 100 -> 128 (4 x 32 K-tiles)
#define NUM_WG   64       // 64 WGs x (4 gates x 16 rows) = 4096 gate rows

typedef __attribute__((ext_vector_type(16))) __bf16 v16bf;
typedef __attribute__((ext_vector_type(8)))  float  v8f;

// ---- workspace layout (byte offsets) ----
#define OFF_CTR   0ull                      // 64 B   : grid-sync counter
#define OFF_H     256ull                    // 4 KB   : h broadcast buffer
#define OFF_C     8192ull                   // 4 KB   : final cell state
#define OFF_BIAS  16384ull                  // 128 KB : b_ih + b_hh, fp32
#define OFF_XBF   (256ull * 1024)           // 1 MB   : x packed bf16 [4096][128]
#define OFF_WIH   (2ull * 1024 * 1024)      // 8 MB   : W_ih bf16 A-tiles
#define OFF_WHH   (16ull * 1024 * 1024)     // 64 MB  : W_hh bf16 A-tiles

static __device__ __forceinline__ float sigmoidf_(float x) {
    return 1.0f / (1.0f + __expf(-x));
}

// A-fragment placement for 16-bit 16x32 A (ISA 7.12.2):
//   lane = m + 16*((k>>3)&1) ;  slot = (k&7) + 8*((k>>4)&1)
// Tile = 512 halves (1 KB): [lane][slot] -> lane*16 + slot.

__global__ void pack_whh_kernel(const float* __restrict__ Whh,
                                __bf16* __restrict__ dst) {
    const unsigned N = (unsigned)LAYERS * 4u * HIDDEN * HIDDEN;  // 2^25
    for (unsigned i = blockIdx.x * blockDim.x + threadIdx.x; i < N;
         i += gridDim.x * blockDim.x) {
        unsigned k   = i & (HIDDEN - 1);
        unsigned row = (i >> 10) & 4095u;
        unsigned l   = i >> 22;
        unsigned tm = row >> 4, m = row & 15u;
        unsigned tk = k >> 5,   kk = k & 31u;
        unsigned lane = m + 16u * ((kk >> 3) & 1u);
        unsigned slot = (kk & 7u) + 8u * ((kk >> 4) & 1u);
        unsigned tile = (l * 256u + tm) * 32u + tk;
        dst[(size_t)tile * 512u + lane * 16u + slot] = (__bf16)Whh[i];
    }
}

__global__ void pack_wih_kernel(const float* __restrict__ Wih,
                                __bf16* __restrict__ dst) {
    const unsigned N = (unsigned)LAYERS * 4u * HIDDEN * KPAD;    // 2^22
    for (unsigned i = blockIdx.x * blockDim.x + threadIdx.x; i < N;
         i += gridDim.x * blockDim.x) {
        unsigned k   = i & (KPAD - 1);
        unsigned row = (i >> 7) & 4095u;
        unsigned l   = i >> 19;
        float v = (k < LETTERS)
                      ? Wih[((size_t)(l * 4096u + row)) * LETTERS + k]
                      : 0.0f;
        unsigned tm = row >> 4, m = row & 15u;
        unsigned tk = k >> 5,   kk = k & 31u;
        unsigned lane = m + 16u * ((kk >> 3) & 1u);
        unsigned slot = (kk & 7u) + 8u * ((kk >> 4) & 1u);
        unsigned tile = (l * 256u + tm) * 4u + tk;
        dst[(size_t)tile * 512u + lane * 16u + slot] = (__bf16)v;
    }
}

__global__ void pack_misc_kernel(const float* __restrict__ b_ih,
                                 const float* __restrict__ b_hh,
                                 const float* __restrict__ website,
                                 const float* __restrict__ payload,
                                 float* __restrict__ bias,
                                 __bf16* __restrict__ xbf) {
    const unsigned N1 = (unsigned)LAYERS * 4u * HIDDEN;          // 32768
    const unsigned N2 = (unsigned)T_TOTAL * KPAD;                // 524288
    for (unsigned i = blockIdx.x * blockDim.x + threadIdx.x; i < N1 + N2;
         i += gridDim.x * blockDim.x) {
        if (i < N1) {
            bias[i] = b_ih[i] + b_hh[i];
        } else {
            unsigned j = i - N1;
            unsigned t = j >> 7, k = j & (KPAD - 1);
            float v = 0.0f;
            if (k < LETTERS)
                v = (t < 2048u) ? website[(size_t)t * LETTERS + k]
                                : payload[(size_t)(t - 2048u) * LETTERS + k];
            xbf[j] = (__bf16)v;
        }
    }
}

__global__ __launch_bounds__(256, 1)
void lstm_persistent_kernel(const __bf16* __restrict__ whh,
                            const __bf16* __restrict__ wih,
                            const float*  __restrict__ bias,
                            const __bf16* __restrict__ xbf,
                            float* __restrict__ h_glob,
                            float* __restrict__ c_glob,
                            unsigned* __restrict__ ctr) {
    __shared__ __bf16 hbf[HIDDEN];        // current hidden state, bf16
    __shared__ __bf16 xtb[KPAD];          // current input row, bf16
    __shared__ float  partial_s[2][4][16]; // [K-half][gate][row] partial sums
    __shared__ float  c_s[16];            // WG-resident cell state slice

    const int tid  = threadIdx.x;
    const int wg   = blockIdx.x;          // owns h indices [16*wg, 16*wg+16)
    const int w    = tid >> 5;            // wave 0..7
    const int lane = tid & 31;
    const int n    = lane & 15;           // column within B/D fragment
    const int g2   = lane >> 4;           // half-select (K / M split)
    const int gb   = w >> 1;              // gate block: 0=i 1=f 2=g 3=o
    const int kh   = w & 1;               // K-half handled by this wave
    const int row_base = gb * HIDDEN + wg * 16;
    const int tile_m   = row_base >> 4;   // = gb*64 + wg

    for (int k2 = tid; k2 < HIDDEN; k2 += 256) hbf[k2] = (__bf16)0.0f;
    if (tid < 16) c_s[tid] = 0.0f;
    __syncthreads();

    // B fragment: column 0 = vec[c0..c0+31]. Address depends only on g2, so
    // every column of B replicates column 0 (finite values) -> no zeroing
    // needed; we only ever read column 0 of D. Two ds_load_b128 per lane.
    auto bfrag = [&](const __bf16* vec, int c0) -> v16bf {
        union { uint4 q[2]; v16bf v; } u;
        const uint4* p = (const uint4*)(vec + c0 + g2 * 8);
        u.q[0] = p[0];
        u.q[1] = p[2];                    // +32 bytes = +16 halves
        return u.v;
    };

    for (int t = 0; t < T_TOTAL; ++t) {
        if (tid < KPAD) xtb[tid] = xbf[(size_t)t * KPAD + tid];
        __syncthreads();

        for (int l = 0; l < LAYERS; ++l) {
            v8f acc0 = {0.f,0.f,0.f,0.f,0.f,0.f,0.f,0.f};
            v8f acc1 = {0.f,0.f,0.f,0.f,0.f,0.f,0.f,0.f};

            // ---- input projection: W_ih (K=128, this wave: 2 K-tiles) ----
            const __bf16* wihL = wih + ((size_t)(l * 256 + tile_m) * 4u) * 512u;
            #pragma unroll
            for (int i = 0; i < 2; ++i) {
                const int tk = kh * 2 + i;
                v16bf a = *(const v16bf*)(wihL + (size_t)tk * 512u + lane * 16);
                v16bf b = bfrag(xtb, tk * 32);
                if (i & 1)
                    acc1 = __builtin_amdgcn_wmma_f32_16x16x32_bf16(
                        false, a, false, b, (short)0, acc1, false, false);
                else
                    acc0 = __builtin_amdgcn_wmma_f32_16x16x32_bf16(
                        false, a, false, b, (short)0, acc0, false, false);
            }

            // ---- recurrent: W_hh (K=1024, this wave: 16 K-tiles) ----
            const __bf16* whhL = whh + ((size_t)(l * 256 + tile_m) * 32u) * 512u;
            #pragma unroll
            for (int i = 0; i < 16; ++i) {
                const int tk = kh * 16 + i;
                if (i + 4 < 16)   // WGP-scope prefetch: pull into near cache
                    __builtin_prefetch(whhL + (size_t)(tk + 4) * 512u + lane * 16, 0, 3);
                v16bf a = *(const v16bf*)(whhL + (size_t)tk * 512u + lane * 16);
                v16bf b = bfrag(hbf, tk * 32);
                if (i & 1)
                    acc1 = __builtin_amdgcn_wmma_f32_16x16x32_bf16(
                        false, a, false, b, (short)0, acc1, false, false);
                else
                    acc0 = __builtin_amdgcn_wmma_f32_16x16x32_bf16(
                        false, a, false, b, (short)0, acc0, false, false);
            }

            // ---- D column 0: lane0 -> rows +0..7, lane16 -> rows +8..15 ----
            if (n == 0) {
                const int rb = g2 * 8;
                #pragma unroll
                for (int r = 0; r < 8; ++r)
                    partial_s[kh][gb][rb + r] = acc0[r] + acc1[r];
            }
            __syncthreads();

            // ---- elementwise cell update (WG-local c), publish new h ----
            if (tid < 16) {
                const float* bp = bias + (size_t)l * 4u * HIDDEN + wg * 16 + tid;
                float ig = sigmoidf_(partial_s[0][0][tid] + partial_s[1][0][tid] + bp[0 * HIDDEN]);
                float fg = sigmoidf_(partial_s[0][1][tid] + partial_s[1][1][tid] + bp[1 * HIDDEN]);
                float gg = tanhf    (partial_s[0][2][tid] + partial_s[1][2][tid] + bp[2 * HIDDEN]);
                float og = sigmoidf_(partial_s[0][3][tid] + partial_s[1][3][tid] + bp[3 * HIDDEN]);
                float c2 = __builtin_fmaf(fg, c_s[tid], ig * gg);
                c_s[tid] = c2;
                float h2 = og * tanhf(c2);
                __hip_atomic_store(&h_glob[wg * 16 + tid], h2,
                                   __ATOMIC_RELAXED, __HIP_MEMORY_SCOPE_AGENT);
                __threadfence();
            }
            __syncthreads();

            // ---- one grid-wide sync per cell (monotonic counter) ----
            if (tid == 0) {
                const unsigned target =
                    ((unsigned)(t * LAYERS + l) + 1u) * (unsigned)NUM_WG;
                __threadfence();
                atomicAdd(ctr, 1u);
                while (__hip_atomic_load(ctr, __ATOMIC_ACQUIRE,
                                         __HIP_MEMORY_SCOPE_AGENT) < target)
                    __builtin_amdgcn_s_sleep(1);
                __threadfence();
            }
            __syncthreads();

            // ---- re-broadcast full h into LDS as bf16 ----
            for (int k2 = tid; k2 < HIDDEN; k2 += 256)
                hbf[k2] = (__bf16)__hip_atomic_load(
                    &h_glob[k2], __ATOMIC_RELAXED, __HIP_MEMORY_SCOPE_AGENT);
            __syncthreads();
        }
    }

    if (tid < 16) c_glob[wg * 16 + tid] = c_s[tid];
}

__global__ void finalize_kernel(const float* __restrict__ c_glob,
                                const float* __restrict__ W_lin,
                                const float* __restrict__ b_lin,
                                const float* __restrict__ W_out,
                                const float* __restrict__ b_out,
                                float* __restrict__ out) {
    __shared__ float red[256];
    __shared__ float feat[16];
    const int tid = threadIdx.x;
    const int r = tid >> 4, j = tid & 15;   // 16 rows x 16 partial lanes
    float s = 0.0f;
    for (int k = j; k < HIDDEN; k += 16)
        s += W_lin[(size_t)r * HIDDEN + k] * c_glob[k];
    red[tid] = s;
    __syncthreads();
    if (j == 0) {
        float tot = b_lin[r];
        for (int q = 0; q < 16; ++q) tot += red[(r << 4) + q];
        feat[r] = tot;
    }
    __syncthreads();
    if (tid == 0) {
        float o = b_out[0];
        for (int q = 0; q < 16; ++q) o += W_out[q] * feat[q];
        out[0] = 1.0f / (1.0f + __expf(-o));
    }
}

extern "C" void kernel_launch(void* const* d_in, const int* in_sizes, int n_in,
                              void* d_out, int out_size, void* d_ws, size_t ws_size,
                              hipStream_t stream) {
    const float* website = (const float*)d_in[0];
    const float* payload = (const float*)d_in[1];
    const float* W_ih    = (const float*)d_in[2];
    const float* W_hh    = (const float*)d_in[3];
    const float* b_ih    = (const float*)d_in[4];
    const float* b_hh    = (const float*)d_in[5];
    const float* W_lin   = (const float*)d_in[6];
    const float* b_lin   = (const float*)d_in[7];
    const float* W_out   = (const float*)d_in[8];
    const float* b_out   = (const float*)d_in[9];

    char* ws = (char*)d_ws;
    unsigned* ctr    = (unsigned*)(ws + OFF_CTR);
    float*    h_glob = (float*)(ws + OFF_H);
    float*    c_glob = (float*)(ws + OFF_C);
    float*    bias   = (float*)(ws + OFF_BIAS);
    __bf16*   xbf    = (__bf16*)(ws + OFF_XBF);
    __bf16*   wih    = (__bf16*)(ws + OFF_WIH);
    __bf16*   whh    = (__bf16*)(ws + OFF_WHH);

    hipMemsetAsync(ctr, 0, 64, stream);

    pack_whh_kernel <<<8192, 256, 0, stream>>>(W_hh, whh);
    pack_wih_kernel <<<2048, 256, 0, stream>>>(W_ih, wih);
    pack_misc_kernel<<<512,  256, 0, stream>>>(b_ih, b_hh, website, payload,
                                               bias, xbf);

    lstm_persistent_kernel<<<NUM_WG, 256, 0, stream>>>(whh, wih, bias, xbf,
                                                       h_glob, c_glob, ctr);

    finalize_kernel<<<1, 256, 0, stream>>>(c_glob, W_lin, b_lin, W_out, b_out,
                                           (float*)d_out);
}
